// MultiHeadAttention_4166118277738
// MI455X (gfx1250) — matrix operand order repaired and checked by reading, not compiled, not run on hardware
//
#include <hip/hip_runtime.h>
#include <hip/hip_bf16.h>

typedef __attribute__((ext_vector_type(16))) __bf16        v16bf;
typedef __attribute__((ext_vector_type(8)))  float         v8f;
typedef __attribute__((ext_vector_type(16))) unsigned short u16x16;
typedef __attribute__((ext_vector_type(8)))  unsigned short u16x8;

#define B_  2
#define T_  2048
#define D_  1024
#define H_  16
#define DK_ 64

// ---------------- helpers ----------------

// native bf16 convert (RNE) -> v_cvt_*bf16_f32, not bit-twiddling
__device__ __forceinline__ unsigned short f2bf(float f) {
  __bf16 h = (__bf16)f;
  return __builtin_bit_cast(unsigned short, h);
}

__device__ __forceinline__ v8f wmma_bf16(v16bf a, v16bf b, v8f c) {
  return __builtin_amdgcn_wmma_f32_16x16x32_bf16(false, a, false, b, (short)0, c,
                                                 false, false);
}

// B-fragment: per-lane contiguous 16 bf16 (lane = column n, K-run of 16 per half)
__device__ __forceinline__ v16bf load16(const unsigned short* p) {
  u16x16 t = *(const u16x16*)p;
  return __builtin_bit_cast(v16bf, t);
}

// A-fragment from a bf16 row. k order used here (contiguous, same as the B-fragments):
// lanes 0-15: elems 0..15 = K kb+0..15
// lanes16-31: elems 0..15 = K kb+16..31
__device__ __forceinline__ v16bf load_a_bf16(const unsigned short* row, int kb, int lane) {
  const unsigned short* p = row + kb + ((lane & 16) ? 16 : 0);
  u16x8 lo = *(const u16x8*)p;
  u16x8 hi = *(const u16x8*)(p + 8);
  u16x16 t;
#pragma unroll
  for (int i = 0; i < 8; ++i) { t[i] = lo[i]; t[8 + i] = hi[i]; }
  return __builtin_bit_cast(v16bf, t);
}

// ---------------- kernel 0: fp32 -> bf16 bulk convert (vectorized) ----------------

__global__ void __launch_bounds__(256) cvt_kernel(const float* __restrict__ src,
                                                  unsigned short* __restrict__ dst,
                                                  int n4) {
  int i = blockIdx.x * blockDim.x + threadIdx.x;
  if (i < n4) {
    float4 f = ((const float4*)src)[i];
    ushort4 o;
    o.x = f2bf(f.x);
    o.y = f2bf(f.y);
    o.z = f2bf(f.z);
    o.w = f2bf(f.w);
    ((ushort4*)dst)[i] = o;
  }
}

// ---------------- kernel 1: fused K/Q/V projection GEMM ----------------
// Each wave computes a 32x32 output tile of X @ W^T + b (M=B*T=4096, N=K=1024).
// Output layout: K,Q -> [B,H,T,DK] bf16 ; V -> [B,H,DK,T] bf16 (transposed).

__device__ __forceinline__ void store_proj(v8f c, int m0, int n0, int lane, int src,
                                           const float* bias, unsigned short* Kp,
                                           unsigned short* Qp, unsigned short* Vt) {
  int ln = lane & 15, hf = lane >> 4;
  int n = n0 + ln;
  float bv = bias[n];
  int h = n >> 6, dk = n & 63;
#pragma unroll
  for (int j = 0; j < 8; ++j) {
    int m = m0 + j + hf * 8;
    int b = m >> 11, t = m & (T_ - 1);
    int bh = b * H_ + h;
    unsigned short bf = f2bf(c[j] + bv);
    if (src == 2)
      Vt[(size_t)(bh * DK_ + dk) * T_ + t] = bf;
    else if (src == 1)
      Qp[((size_t)bh * T_ + t) * DK_ + dk] = bf;
    else
      Kp[((size_t)bh * T_ + t) * DK_ + dk] = bf;
  }
}

__global__ void __launch_bounds__(128) proj_kernel(
    const unsigned short* __restrict__ Kbf, const unsigned short* __restrict__ Qbf,
    const unsigned short* __restrict__ Vbf, const unsigned short* __restrict__ Wbf,
    const float* __restrict__ bias, unsigned short* __restrict__ Kp,
    unsigned short* __restrict__ Qp, unsigned short* __restrict__ Vt) {
  int lane  = threadIdx.x & 31;
  int gw    = blockIdx.x * 4 + (threadIdx.x >> 5);
  int ntile = gw & 31;           // N/32
  int mtile = (gw >> 5) & 127;   // M/32
  int src   = gw >> 12;          // 0=k,1=q,2=v
  const unsigned short* X = (src == 0) ? Kbf : (src == 1) ? Qbf : Vbf;
  int m0 = mtile * 32, n0 = ntile * 32;
  int ln = lane & 15, hf = lane >> 4;

  v8f a00 = {}, a01 = {}, a10 = {}, a11 = {};
  const unsigned short* rowA0 = X + (size_t)(m0 + ln) * D_;
  const unsigned short* rowA1 = X + (size_t)(m0 + 16 + ln) * D_;
  const unsigned short* rowB0 = Wbf + (size_t)(n0 + ln) * D_ + hf * 16;
  const unsigned short* rowB1 = Wbf + (size_t)(n0 + 16 + ln) * D_ + hf * 16;

  for (int kb = 0; kb < D_; kb += 32) {
    __builtin_prefetch(rowA0 + kb + 256, 0, 1); // global_prefetch_b8
    v16bf aA = load_a_bf16(rowA0, kb, lane);
    v16bf aB = load_a_bf16(rowA1, kb, lane);
    v16bf bA = load16(rowB0 + kb);
    v16bf bB = load16(rowB1 + kb);
    a00 = wmma_bf16(aA, bA, a00);
    a01 = wmma_bf16(aA, bB, a01);
    a10 = wmma_bf16(aB, bA, a10);
    a11 = wmma_bf16(aB, bB, a11);
  }
  store_proj(a00, m0,      n0,      lane, src, bias, Kp, Qp, Vt);
  store_proj(a01, m0,      n0 + 16, lane, src, bias, Kp, Qp, Vt);
  store_proj(a10, m0 + 16, n0,      lane, src, bias, Kp, Qp, Vt);
  store_proj(a11, m0 + 16, n0 + 16, lane, src, bias, Kp, Qp, Vt);
}

// ---------------- kernel 2: flash attention ----------------
// One wave per (b,h, 16-query tile). Key tiles of 32 (= one WMMA K step).

__global__ void __launch_bounds__(128) attn_kernel(
    const unsigned short* __restrict__ Qp, const unsigned short* __restrict__ Kp,
    const unsigned short* __restrict__ Vt, const int* __restrict__ mask,
    const int* __restrict__ causalPtr, unsigned short* __restrict__ AttnOut) {
  __shared__ __align__(32) unsigned short Plds[4][16][40]; // padded rows: bank stagger
  int lane = threadIdx.x & 31;
  int wv   = threadIdx.x >> 5;
  int gw   = blockIdx.x * 4 + wv;
  int qt   = gw & 127;   // T/16 query tiles
  int bh   = gw >> 7;    // 0..B*H-1
  int b = bh >> 4, h = bh & 15;
  int ln = lane & 15, hf = lane >> 4;
  int causal = causalPtr[0];

  // Q A-fragments (reused across all key tiles)
  const unsigned short* qrow = Qp + ((size_t)bh * T_ + qt * 16 + ln) * DK_;
  v16bf aq0 = load_a_bf16(qrow, 0, lane);
  v16bf aq1 = load_a_bf16(qrow, 32, lane);

  v8f o[4] = {};
  float mrow[8], lrow[8];
#pragma unroll
  for (int j = 0; j < 8; ++j) { mrow[j] = -__builtin_inff(); lrow[j] = 0.f; }

  const float c1 = 0.125f * 1.44269504088896340736f; // 1/sqrt(DK) * log2(e)
  int ktMax = causal ? ((qt * 16 + 15) >> 5) : (T_ / 32 - 1);
  const int* mp = mask + b * T_;

  for (int kt = 0; kt <= ktMax; ++kt) {
    int key0 = kt * 32;
    const unsigned short* kb0 = Kp + ((size_t)bh * T_ + key0) * DK_;

    // S = Q K^T  (16x32 tile: two 16x16 C-fragments)
    v8f s0 = {}, s1 = {};
    {
      const unsigned short* r = kb0 + (size_t)ln * DK_ + hf * 16;
      s0 = wmma_bf16(aq0, load16(r), s0);
      s0 = wmma_bf16(aq1, load16(r + 32), s0);
      const unsigned short* r2 = kb0 + (size_t)(16 + ln) * DK_ + hf * 16;
      s1 = wmma_bf16(aq0, load16(r2), s1);
      s1 = wmma_bf16(aq1, load16(r2 + 32), s1);
    }
    int keyA = key0 + ln, keyB = key0 + 16 + ln;
    bool okA = mp[keyA] != 0;
    bool okB = mp[keyB] != 0;

    float alpha[8];
#pragma unroll
    for (int j = 0; j < 8; ++j) {
      int q = qt * 16 + j + hf * 8;
      float x0 = (okA && !(causal && keyA > q)) ? s0[j] * c1 : -__builtin_inff();
      float x1 = (okB && !(causal && keyB > q)) ? s1[j] * c1 : -__builtin_inff();
      // row max across the 16 lanes of this half-wave
      float r = fmaxf(x0, x1);
      r = fmaxf(r, __shfl_xor(r, 1, 32));
      r = fmaxf(r, __shfl_xor(r, 2, 32));
      r = fmaxf(r, __shfl_xor(r, 4, 32));
      r = fmaxf(r, __shfl_xor(r, 8, 32));
      float mn   = fmaxf(mrow[j], r);
      bool dead  = (mn == -__builtin_inff());
      float al   = dead ? 1.f : exp2f(mrow[j] - mn);
      float e0   = dead ? 0.f : exp2f(x0 - mn);
      float e1   = dead ? 0.f : exp2f(x1 - mn);
      float ps = e0 + e1;
      ps += __shfl_xor(ps, 1, 32);
      ps += __shfl_xor(ps, 2, 32);
      ps += __shfl_xor(ps, 4, 32);
      ps += __shfl_xor(ps, 8, 32);
      lrow[j] = lrow[j] * al + ps;
      mrow[j] = mn;
      alpha[j] = al;
      // C-layout -> LDS (row = j + 8*half, col = lane%16 per fragment)
      Plds[wv][j + hf * 8][ln]      = f2bf(e0);
      Plds[wv][j + hf * 8][16 + ln] = f2bf(e1);
    }
#pragma unroll
    for (int t = 0; t < 4; ++t)
#pragma unroll
      for (int j = 0; j < 8; ++j) o[t][j] *= alpha[j];

    // read P back as an A-fragment (16x32, K = 32 keys)
    v16bf ap = load_a_bf16(&Plds[wv][ln][0], 0, lane);

    // O += P @ V ; V stored transposed [B,H,DK,T] so B-frag columns are contiguous
#pragma unroll
    for (int t = 0; t < 4; ++t) {
      const unsigned short* vp =
          Vt + ((size_t)bh * DK_ + t * 16 + ln) * T_ + key0 + hf * 16;
      o[t] = wmma_bf16(ap, load16(vp), o[t]);
    }
  }

  // normalize and store concat [B,T,D] bf16
  unsigned short* ob = AttnOut + ((size_t)b * T_ + qt * 16) * D_ + h * DK_;
#pragma unroll
  for (int j = 0; j < 8; ++j) {
    float inv = (lrow[j] > 0.f) ? 1.f / lrow[j] : 0.f;
    unsigned short* orow = ob + (size_t)(j + hf * 8) * D_;
#pragma unroll
    for (int t = 0; t < 4; ++t) orow[t * 16 + ln] = f2bf(o[t][j] * inv);
  }
}

// ---------------- kernel 3: output projection (bf16 A, fp32 out) ----------------

__global__ void __launch_bounds__(128) outproj_kernel(
    const unsigned short* __restrict__ A, const unsigned short* __restrict__ Wbf,
    const float* __restrict__ bias, float* __restrict__ out) {
  int lane  = threadIdx.x & 31;
  int gw    = blockIdx.x * 4 + (threadIdx.x >> 5);
  int ntile = gw & 31;
  int mtile = gw >> 5;  // 0..127
  int m0 = mtile * 32, n0 = ntile * 32;
  int ln = lane & 15, hf = lane >> 4;

  v8f a00 = {}, a01 = {}, a10 = {}, a11 = {};
  const unsigned short* rowA0 = A + (size_t)(m0 + ln) * D_;
  const unsigned short* rowA1 = A + (size_t)(m0 + 16 + ln) * D_;
  const unsigned short* rowB0 = Wbf + (size_t)(n0 + ln) * D_ + hf * 16;
  const unsigned short* rowB1 = Wbf + (size_t)(n0 + 16 + ln) * D_ + hf * 16;

  for (int kb = 0; kb < D_; kb += 32) {
    __builtin_prefetch(rowA0 + kb + 256, 0, 1);
    v16bf aA = load_a_bf16(rowA0, kb, lane);
    v16bf aB = load_a_bf16(rowA1, kb, lane);
    v16bf bA = load16(rowB0 + kb);
    v16bf bB = load16(rowB1 + kb);
    a00 = wmma_bf16(aA, bA, a00);
    a01 = wmma_bf16(aA, bB, a01);
    a10 = wmma_bf16(aB, bA, a10);
    a11 = wmma_bf16(aB, bB, a11);
  }
#pragma unroll
  for (int fi = 0; fi < 4; ++fi) {
    v8f c  = (fi == 0) ? a00 : (fi == 1) ? a01 : (fi == 2) ? a10 : a11;
    int mm = m0 + ((fi >= 2) ? 16 : 0);
    int nn = n0 + ((fi & 1) ? 16 : 0) + ln;
    float bv = bias[nn];
#pragma unroll
    for (int j = 0; j < 8; ++j) {
      int m = mm + j + hf * 8;
      out[(size_t)m * D_ + nn] = c[j] + bv;
    }
  }
}

// ---------------- launch ----------------

extern "C" void kernel_launch(void* const* d_in, const int* in_sizes, int n_in,
                              void* d_out, int out_size, void* d_ws, size_t ws_size,
                              hipStream_t stream) {
  (void)in_sizes; (void)n_in; (void)out_size; (void)ws_size;
  const float* k_in   = (const float*)d_in[0];
  const float* q_in   = (const float*)d_in[1];
  const float* v_in   = (const float*)d_in[2];
  const int*   mask   = (const int*)d_in[3];
  const int*   causal = (const int*)d_in[4];
  const float* W      = (const float*)d_in[5];
  const float* bias   = (const float*)d_in[6];
  float* out = (float*)d_out;

  char* ws = (char*)d_ws;
  unsigned short* Wbf  = (unsigned short*)(ws);                              // 2 MB
  unsigned short* Kbf  = (unsigned short*)(ws + (size_t)2  * 1024 * 1024);   // 8 MB
  unsigned short* Qbf  = (unsigned short*)(ws + (size_t)10 * 1024 * 1024);   // 8 MB
  unsigned short* Vbf  = (unsigned short*)(ws + (size_t)18 * 1024 * 1024);   // 8 MB
  unsigned short* Kp   = (unsigned short*)(ws + (size_t)26 * 1024 * 1024);   // 8 MB
  unsigned short* Qp   = (unsigned short*)(ws + (size_t)34 * 1024 * 1024);   // 8 MB
  unsigned short* Vt   = (unsigned short*)(ws + (size_t)42 * 1024 * 1024);   // 8 MB
  unsigned short* Attn = (unsigned short*)(ws + (size_t)50 * 1024 * 1024);   // 8 MB

  const int nW = D_ * D_ / 4;       // float4 groups in W
  const int nX = B_ * T_ * D_ / 4;  // float4 groups per input
  hipLaunchKernelGGL(cvt_kernel, dim3((nW + 255) / 256), dim3(256), 0, stream,
                     W, Wbf, nW);
  hipLaunchKernelGGL(cvt_kernel, dim3((nX + 255) / 256), dim3(256), 0, stream,
                     k_in, Kbf, nX);
  hipLaunchKernelGGL(cvt_kernel, dim3((nX + 255) / 256), dim3(256), 0, stream,
                     q_in, Qbf, nX);
  hipLaunchKernelGGL(cvt_kernel, dim3((nX + 255) / 256), dim3(256), 0, stream,
                     v_in, Vbf, nX);

  hipLaunchKernelGGL(proj_kernel, dim3(3 * 128 * 32 / 4), dim3(128), 0, stream,
                     Kbf, Qbf, Vbf, Wbf, bias, Kp, Qp, Vt);
  hipLaunchKernelGGL(attn_kernel, dim3(B_ * H_ * (T_ / 16) / 4), dim3(128), 0, stream,
                     Qp, Kp, Vt, mask, causal, Attn);
  hipLaunchKernelGGL(outproj_kernel, dim3(128 * 32 / 4), dim3(128), 0, stream,
                     Attn, Wbf, bias, out);
}
